// attentive_node_features_dag_1486058684456
// MI455X (gfx1250) — matrix operand order repaired and checked by reading, not compiled, not run on hardware
//
#include <hip/hip_runtime.h>

// ---------------------------------------------------------------------------
// attentive node features — f32-via-bf16x3 split WMMA (v_wmma_f32_16x16x32_bf16)
// + Tensor Data Mover staging of key tiles.
// B=16, N=2048, V=1024, f32 in/out.
// Split: x = hi + lo (both bf16; exact since bf16 exponent == f32 exponent),
// product emulated as Ah*Bh + Ah*Bl + Al*Bh (f32 accumulate) -> ~16 mantissa
// bits, 3 WMMAs per K=32 chunk vs 8 f32 16x16x4 WMMAs (2.67x fewer issues).
// Softmax identity: alpha*mask/sum(alpha*mask) == exp(tanh(s))*mask / sum(...).
// ---------------------------------------------------------------------------

typedef float v8f __attribute__((ext_vector_type(8)));
typedef __bf16 v16bf __attribute__((ext_vector_type(16)));
typedef __bf16 v8bf __attribute__((ext_vector_type(8)));
typedef __bf16 v4bf __attribute__((ext_vector_type(4)));
typedef unsigned int u32x4 __attribute__((ext_vector_type(4)));
typedef int i32x4 __attribute__((ext_vector_type(4)));
typedef int i32x8 __attribute__((ext_vector_type(8)));

#define WMMA_BF16(A, Bf, C)                                                    \
  __builtin_amdgcn_wmma_f32_16x16x32_bf16(false, (A), false, (Bf), (short)0,   \
                                          (C), false, false)

static constexpr int Bb = 16;
static constexpr int Nn = 2048;
static constexpr int Vv = 1024;

__device__ __forceinline__ v16bf cat8(v8bf a, v8bf b) {
  return __builtin_shufflevector(a, b, 0, 1, 2, 3, 4, 5, 6, 7, 8, 9, 10, 11,
                                 12, 13, 14, 15);
}

// Split 16 consecutive f32 (16B-aligned) into bf16 hi/lo fragments.
__device__ __forceinline__ void split16(const float* s, v16bf& h, v16bf& l) {
  const float4* s4 = (const float4*)s;
#pragma unroll
  for (int q = 0; q < 4; ++q) {
    float4 v = s4[q];
    float xs[4] = {v.x, v.y, v.z, v.w};
#pragma unroll
    for (int e = 0; e < 4; ++e) {
      float x = xs[e];
      __bf16 hh = (__bf16)x;
      h[q * 4 + e] = hh;
      l[q * 4 + e] = (__bf16)(x - (float)hh);
    }
  }
}

// ===========================================================================
// TDM: DMA a rows x 1024 f32 tile (row stride 1024) from global into LDS.
// ===========================================================================
__device__ __forceinline__ void tdm_load_tile(const float* gsrc, float* ldsdst,
                                              unsigned rows) {
  unsigned long long ga = (unsigned long long)(const void*)gsrc;
  unsigned lds = (unsigned)(unsigned long long)(void*)ldsdst;

  u32x4 g0;
  g0[0] = 1u;                                  // count=1
  g0[1] = lds;                                 // lds_addr
  g0[2] = (unsigned)(ga & 0xFFFFFFFFu);        // global_addr lo
  g0[3] = (unsigned)((ga >> 32) & 0x1FFFFFFu) | (2u << 30);  // addr hi | type=2

  i32x8 g1;
  g1[0] = (int)(2u << 16);     // data_size=2 (4 bytes)
  g1[1] = (int)(1024u << 16);  // tensor_dim0 = 1024
  g1[2] = (int)(rows << 16);   // tensor_dim1 = rows
  g1[3] = (int)(1024u << 16);  // tile_dim0 = 1024
  g1[4] = (int)rows;           // tile_dim1 = rows
  g1[5] = 1024;                // tensor_dim0_stride
  g1[6] = 0;
  g1[7] = 0;

  i32x4 gz = {0, 0, 0, 0};
#if defined(__clang_major__) && __clang_major__ >= 23
  i32x8 gz8 = {0, 0, 0, 0, 0, 0, 0, 0};
  __builtin_amdgcn_tensor_load_to_lds(g0, g1, gz, gz, gz8, 0);
#else
  __builtin_amdgcn_tensor_load_to_lds(g0, g1, gz, gz, 0);
#endif
}

// ===========================================================================
// Kernel 0: split W (V x V f32, row-major [o][v]) into bf16 hi/lo planes.
// ===========================================================================
__global__ __launch_bounds__(256) void prep_w(const float* __restrict__ W,
                                              __bf16* __restrict__ Wh,
                                              __bf16* __restrict__ Wl) {
  const int idx = blockIdx.x * 256 + threadIdx.x;  // float4 index
  float4 v = ((const float4*)W)[idx];
  float xs[4] = {v.x, v.y, v.z, v.w};
  v4bf h, l;
#pragma unroll
  for (int e = 0; e < 4; ++e) {
    __bf16 hh = (__bf16)xs[e];
    h[e] = hh;
    l[e] = (__bf16)(xs[e] - (float)hh);
  }
  ((v4bf*)Wh)[idx] = h;
  ((v4bf*)Wl)[idx] = l;
}

// ===========================================================================
// Kernel 1: X = F @ W^T + bias, written as bf16 hi/lo planes.
// Block: 256 threads (8 waves), 32 rows x 1024 cols.
// ===========================================================================
__global__ __launch_bounds__(256) void proj_kernel(
    const float* __restrict__ F, const __bf16* __restrict__ Wh,
    const __bf16* __restrict__ Wl, const float* __restrict__ bias,
    __bf16* __restrict__ Xh, __bf16* __restrict__ Xl) {
  __shared__ __bf16 sFH[32 * Vv];  // 64 KB
  __shared__ __bf16 sFL[32 * Vv];  // 64 KB

  const int tid = threadIdx.x;
  const int wave = tid >> 5;
  const int lane = tid & 31;
  const int l16 = lane & 15;
  const int hi = lane >> 4;
  const long r0 = (long)blockIdx.x * 32;

  // Stage + split the 32x1024 F tile.
  {
    const float4* src = (const float4*)(F + r0 * Vv);
    for (int idx = tid; idx < 32 * Vv / 4; idx += 256) {
      float4 v = src[idx];
      float xs[4] = {v.x, v.y, v.z, v.w};
      v4bf h, l;
#pragma unroll
      for (int e = 0; e < 4; ++e) {
        __bf16 hh = (__bf16)xs[e];
        h[e] = hh;
        l[e] = (__bf16)(xs[e] - (float)hh);
      }
      ((v4bf*)sFH)[idx] = h;
      ((v4bf*)sFL)[idx] = l;
    }
  }
  __syncthreads();

  for (int pass = 0; pass < 2; ++pass) {
    const int cbase = pass * 512 + wave * 64;
    v8f acc[2][4];
#pragma unroll
    for (int rt = 0; rt < 2; ++rt)
#pragma unroll
      for (int ct = 0; ct < 4; ++ct)
        acc[rt][ct] = (v8f){0.f, 0.f, 0.f, 0.f, 0.f, 0.f, 0.f, 0.f};

    for (int k0 = 0; k0 < Vv; k0 += 32) {
      // A fragments: element e = K (e + 16*hi)
      v16bf aH[2], aL[2];
#pragma unroll
      for (int rt = 0; rt < 2; ++rt) {
        const int base = (rt * 16 + l16) * Vv + k0;
        aH[rt] = cat8(*(const v8bf*)&sFH[base + 16 * hi],
                      *(const v8bf*)&sFH[base + 16 * hi + 8]);
        aL[rt] = cat8(*(const v8bf*)&sFL[base + 16 * hi],
                      *(const v8bf*)&sFL[base + 16 * hi + 8]);
      }
#pragma unroll
      for (int ct = 0; ct < 4; ++ct) {
        const int c = cbase + ct * 16 + l16;
        const __bf16* wrh = Wh + (long)c * Vv + k0 + 16 * hi;
        const __bf16* wrl = Wl + (long)c * Vv + k0 + 16 * hi;
        // B fragment: element e = K (e + 16*hi)
        v16bf bH = cat8(*(const v8bf*)&wrh[0], *(const v8bf*)&wrh[8]);
        v16bf bL = cat8(*(const v8bf*)&wrl[0], *(const v8bf*)&wrl[8]);
#pragma unroll
        for (int rt = 0; rt < 2; ++rt) {
          acc[rt][ct] = WMMA_BF16(aH[rt], bH, acc[rt][ct]);
          acc[rt][ct] = WMMA_BF16(aH[rt], bL, acc[rt][ct]);
          acc[rt][ct] = WMMA_BF16(aL[rt], bH, acc[rt][ct]);
        }
      }
    }

#pragma unroll
    for (int ct = 0; ct < 4; ++ct) {
      const int c = cbase + ct * 16 + l16;
      const float bb = bias[c];
#pragma unroll
      for (int rt = 0; rt < 2; ++rt) {
#pragma unroll
        for (int r = 0; r < 8; ++r) {
          const long row = r0 + rt * 16 + r + 8 * hi;
          const float x = acc[rt][ct][r] + bb;
          const __bf16 h = (__bf16)x;
          Xh[row * Vv + c] = h;
          Xl[row * Vv + c] = (__bf16)(x - (float)h);
        }
      }
    }
  }
}

// ===========================================================================
// Kernel 2: fused scores/softmax/pool, 32-key tiles, bf16x3 everywhere.
// One block per (batch, 16-query tile); 8 waves.
// ===========================================================================
__global__ __launch_bounds__(256) void attn_kernel(
    const float* __restrict__ F, const int* __restrict__ lengths,
    const __bf16* __restrict__ Xh, const __bf16* __restrict__ Xl,
    float* __restrict__ out) {
  __shared__ float sRaw[32 * Vv];     // 128 KB  TDM-staged f32 key tile
  __shared__ __bf16 sTrH[Vv * 32];    // 64 KB   [col][key] hi
  __shared__ __bf16 sTrL[Vv * 32];    // 64 KB   [col][key] lo
  __shared__ float sPart[16 * 256];   // 16 KB   S partials [wave*2+st]
  __shared__ float sP[512];           // P (f32, for denominator)
  __shared__ __bf16 sPH[512];         // P hi
  __shared__ __bf16 sPL[512];         // P lo
  __shared__ float sDen[16];

  const int tid = threadIdx.x;
  const int wave = tid >> 5;
  const int lane = tid & 31;
  const int l16 = lane & 15;
  const int hi = lane >> 4;

  const int b = blockIdx.x >> 7;
  const int qt = blockIdx.x & 127;
  const int qbase = qt * 16;
  const int len = lengths[b];
  const float* Fb = F + (long)b * Nn * Vv;

  const int ktlen = (len - 1) >> 4;
  const int ktmax16 = qt < ktlen ? qt : ktlen;
  const int ndt = (ktmax16 >> 1) + 1;  // number of 32-key tiles

  if (tid == 0) tdm_load_tile(Fb, sRaw, 32u);

  // Hoist X A-fragments (16 rows x this wave's 128-d slice) into registers.
  v16bf aXH[4], aXL[4];
  {
    const long rowo = ((long)b * Nn + qbase + l16) * Vv + wave * 128;
#pragma unroll
    for (int ch = 0; ch < 4; ++ch) {
      const long base = rowo + ch * 32;
      aXH[ch] = cat8(*(const v8bf*)&Xh[base + 16 * hi],
                     *(const v8bf*)&Xh[base + 16 * hi + 8]);
      aXL[ch] = cat8(*(const v8bf*)&Xl[base + 16 * hi],
                     *(const v8bf*)&Xl[base + 16 * hi + 8]);
    }
  }

  if (tid < 16) sDen[tid] = 0.f;

  v8f acc[8];
#pragma unroll
  for (int ct = 0; ct < 8; ++ct)
    acc[ct] = (v8f){0.f, 0.f, 0.f, 0.f, 0.f, 0.f, 0.f, 0.f};

  for (int dt = 0; dt < ndt; ++dt) {
    const int kb32 = dt * 32;
    __builtin_amdgcn_s_wait_tensorcnt((short)0);
    __syncthreads();  // raw tile visible

    // --- Build key-contiguous transposed bf16 hi/lo (for the numerator) ---
    // Thread t owns cols 4t..4t+3; pack 8 keys per b128 store.
#pragma unroll
    for (int kb = 0; kb < 4; ++kb) {
      float4 rowv[8];
#pragma unroll
      for (int kk = 0; kk < 8; ++kk)
        rowv[kk] = *(const float4*)&sRaw[(kb * 8 + kk) * Vv + tid * 4];
#pragma unroll
      for (int cc = 0; cc < 4; ++cc) {
        v8bf h, l;
#pragma unroll
        for (int kk = 0; kk < 8; ++kk) {
          float x = (cc == 0) ? rowv[kk].x
                  : (cc == 1) ? rowv[kk].y
                  : (cc == 2) ? rowv[kk].z
                              : rowv[kk].w;
          __bf16 hh = (__bf16)x;
          h[kk] = hh;
          l[kk] = (__bf16)(x - (float)hh);
        }
        *(v8bf*)&sTrH[(tid * 4 + cc) * 32 + kb * 8] = h;
        *(v8bf*)&sTrL[(tid * 4 + cc) * 32 + kb * 8] = l;
      }
    }
    __syncthreads();

    // --- Scores: S[st] = Xq @ Fk^T over this wave's 128-d slice ---
#pragma unroll
    for (int st = 0; st < 2; ++st) {
      v8f s = (v8f){0.f, 0.f, 0.f, 0.f, 0.f, 0.f, 0.f, 0.f};
#pragma unroll
      for (int ch = 0; ch < 4; ++ch) {
        const float* src =
            &sRaw[(st * 16 + l16) * Vv + wave * 128 + ch * 32 + 16 * hi];
        v16bf bH, bL;
        split16(src, bH, bL);
        s = WMMA_BF16(aXH[ch], bH, s);
        s = WMMA_BF16(aXH[ch], bL, s);
        s = WMMA_BF16(aXL[ch], bH, s);
      }
#pragma unroll
      for (int r = 0; r < 8; ++r)
        sPart[(wave * 2 + st) * 256 + (r + 8 * hi) * 16 + l16] = s[r];
    }
    __syncthreads();  // all raw reads done

    // Overlap next tile's DMA with P-transform + numerator.
    if (dt + 1 < ndt && tid == 0)
      tdm_load_tile(Fb + (long)(dt + 1) * 32 * Vv, sRaw, 32u);

    // --- Reduce partials (fixed order), mask + exp(tanh), split P ---
#pragma unroll
    for (int e0 = 0; e0 < 2; ++e0) {
      const int e = tid + e0 * 256;
      const int i = e >> 5;
      const int j = e & 31;
      float v = 0.f;
#pragma unroll
      for (int w = 0; w < 8; ++w)
        v += sPart[(w * 2 + (j >> 4)) * 256 + i * 16 + (j & 15)];
      const int gm = kb32 + j;
      const bool ok = (gm <= qbase + i) && (gm < len);
      const float p = ok ? __expf(tanhf(v)) : 0.f;
      sP[e] = p;
      const __bf16 h = (__bf16)p;
      sPH[e] = h;
      sPL[e] = (__bf16)(p - (float)h);
    }
    __syncthreads();

    // Row denominators (16 threads, fixed order => deterministic).
    if (tid < 16) {
      float s = 0.f;
#pragma unroll
      for (int j = 0; j < 32; ++j) s += sP[tid * 32 + j];
      sDen[tid] += s;
    }

    // --- Numerator: acc += P @ Fk for this wave's 128 columns ---
    const v16bf aPH = cat8(*(const v8bf*)&sPH[l16 * 32 + 16 * hi],
                           *(const v8bf*)&sPH[l16 * 32 + 16 * hi + 8]);
    const v16bf aPL = cat8(*(const v8bf*)&sPL[l16 * 32 + 16 * hi],
                           *(const v8bf*)&sPL[l16 * 32 + 16 * hi + 8]);
#pragma unroll
    for (int ct = 0; ct < 8; ++ct) {
      const int c = wave * 128 + ct * 16 + l16;
      v16bf bH = cat8(*(const v8bf*)&sTrH[c * 32 + 16 * hi],
                      *(const v8bf*)&sTrH[c * 32 + 16 * hi + 8]);
      v16bf bL = cat8(*(const v8bf*)&sTrL[c * 32 + 16 * hi],
                      *(const v8bf*)&sTrL[c * 32 + 16 * hi + 8]);
      acc[ct] = WMMA_BF16(aPH, bH, acc[ct]);
      acc[ct] = WMMA_BF16(aPH, bL, acc[ct]);
      acc[ct] = WMMA_BF16(aPL, bH, acc[ct]);
    }
    __syncthreads();  // sTr / sP readers done before next overwrite
  }

  // --- Epilogue ---
  float inv[8];
#pragma unroll
  for (int r = 0; r < 8; ++r) inv[r] = 1.f / sDen[r + 8 * hi];

  float* outB = out + ((long)b * Nn + qbase) * Vv;
#pragma unroll
  for (int ct = 0; ct < 8; ++ct) {
    const int c = wave * 128 + ct * 16 + l16;
#pragma unroll
    for (int r = 0; r < 8; ++r)
      outB[(long)(r + 8 * hi) * Vv + c] = acc[ct][r] * inv[r];
  }
}

// ===========================================================================
extern "C" void kernel_launch(void* const* d_in, const int* in_sizes, int n_in,
                              void* d_out, int out_size, void* d_ws,
                              size_t ws_size, hipStream_t stream) {
  const float* F = (const float*)d_in[0];      // (B, N, V) f32
  const int* lengths = (const int*)d_in[1];    // (B,) i32
  const float* W = (const float*)d_in[2];      // (V, V) f32
  const float* bias = (const float*)d_in[3];   // (V,) f32
  float* out = (float*)d_out;

  // Workspace layout (132 MiB): X hi/lo planes + W hi/lo planes.
  __bf16* Xh = (__bf16*)d_ws;
  __bf16* Xl = Xh + (size_t)Bb * Nn * Vv;
  __bf16* Wh = Xl + (size_t)Bb * Nn * Vv;
  __bf16* Wl = Wh + (size_t)Vv * Vv;

  prep_w<<<(Vv * Vv / 4) / 256, 256, 0, stream>>>(W, Wh, Wl);
  proj_kernel<<<(Bb * Nn) / 32, 256, 0, stream>>>(F, Wh, Wl, bias, Xh, Xl);
  attn_kernel<<<Bb * (Nn / 16), 256, 0, stream>>>(F, lengths, Xh, Xl, out);
}
